// LSTM_25512105738298
// MI455X (gfx1250) — compile-verified, matrix operand order checked by reading, not run on hardware
//
#include <hip/hip_runtime.h>
#include <hip/hip_bf16.h>

typedef __attribute__((ext_vector_type(16))) __bf16 v16bf;
typedef __attribute__((ext_vector_type(8)))  float  v8f;
typedef __attribute__((ext_vector_type(4)))  __bf16 bf4;

#define I_DIM 1024
#define H_DIM 2048
#define O_DIM 50257
#define B_DIM 1024
#define G_DIM (4 * H_DIM)   // 8192
#define O_PAD 50304         // 393 * 128 (padded so N-tiles need no load guards)

#define MT 128
#define NT 128
#define KT 64

// ---------------------------------------------------------------------------
// f32 -> bf16 conversion with zero tail padding (vectorized x4).
// n_src and n_dst are both multiples of 4.
// ---------------------------------------------------------------------------
__global__ void cvt_f32_to_bf16(const float* __restrict__ src, __bf16* __restrict__ dst,
                                long n_src, long n_dst) {
  long i = ((long)blockIdx.x * blockDim.x + threadIdx.x) * 4;
  if (i >= n_dst) return;
  bf4 v;
  if (i < n_src) {
    const float4 f = *reinterpret_cast<const float4*>(src + i);
    v.x = (__bf16)f.x; v.y = (__bf16)f.y; v.z = (__bf16)f.z; v.w = (__bf16)f.w;
  } else {
    v.x = (__bf16)0.f; v.y = (__bf16)0.f; v.z = (__bf16)0.f; v.w = (__bf16)0.f;
  }
  *reinterpret_cast<bf4*>(dst + i) = v;
}

// ---------------------------------------------------------------------------
// C[M,N] = A1[M,K1]*W1[N,K1]^T (+ A2[M,K2]*W2[N,K2]^T) + bias1[n] (+ bias2[n])
// Both operands row-major with K contiguous -> WMMA A and B fragments are each
// 16 contiguous bf16 per lane (lane%16 = row, (lane/16)*16 = k offset).
// Block: 256 threads = 8 waves, 128x128 tile; wave grid 2(M) x 4(N),
// each wave 64x32 = 4x2 WMMA accumulators.
// Staging: double-buffered GLOBAL_LOAD_ASYNC_TO_LDS_B128 (ASYNCcnt-tracked,
// no VGPR round-trip), prefetching stage s+1 while computing stage s.
// ---------------------------------------------------------------------------
__global__ __launch_bounds__(256, 1)
void gemm_bf16_nt(const __bf16* __restrict__ A1, const __bf16* __restrict__ W1, int K1,
                  const __bf16* __restrict__ A2, const __bf16* __restrict__ W2, int K2,
                  const float* __restrict__ bias1, const float* __restrict__ bias2,
                  float* __restrict__ C, long ldc, int nvalid) {
  __shared__ __align__(32) __bf16 As[2][MT * KT];
  __shared__ __align__(32) __bf16 Bs[2][NT * KT];

  const int tid  = threadIdx.x;
  const int lane = tid & 31;
  const int wave = tid >> 5;
  const int wm   = wave & 1;   // 0..1 -> 64-row strip
  const int wn   = wave >> 1;  // 0..3 -> 32-col strip
  const int m0   = blockIdx.x * MT;
  const int n0   = blockIdx.y * NT;

  const int lrow = lane & 15;        // fragment row (A) / W row (B)
  const int lko  = (lane >> 4) * 16; // fragment K offset

  const int s1  = K1 / KT;
  const int nst = s1 + K2 / KT;

  v8f acc[4][2] = {};

  // Issue one k-stage of async global->LDS copies: 256 threads x 4 x 16B
  // for each of A and W (8 async b128 per thread per stage).
  auto issue_stage = [&](int s, int buf) {
    const __bf16* A;
    const __bf16* W;
    int K, k0;
    if (s < s1) { A = A1; W = W1; K = K1; k0 = s * KT; }
    else        { A = A2; W = W2; K = K2; k0 = (s - s1) * KT; }
    #pragma unroll
    for (int v = 0; v < 4; ++v) {
      const int vec = tid + v * 256;       // 0..1023
      const int row = vec >> 3;            // KT/8 = 8 vectors per row
      const int col = (vec & 7) * 8;
      // LDS byte offset = low 32 bits of the generic (flat) LDS address.
      const unsigned la = (unsigned)(size_t)&As[buf][row * KT + col];
      const unsigned lb = (unsigned)(size_t)&Bs[buf][row * KT + col];
      const unsigned long long ga =
          (unsigned long long)(size_t)&A[(long)(m0 + row) * K + k0 + col];
      const unsigned long long gb =
          (unsigned long long)(size_t)&W[(long)(n0 + row) * K + k0 + col];
      asm volatile("global_load_async_to_lds_b128 %0, %1, off"
                   :: "v"(la), "v"(ga) : "memory");
      asm volatile("global_load_async_to_lds_b128 %0, %1, off"
                   :: "v"(lb), "v"(gb) : "memory");
    }
  };

  issue_stage(0, 0);

  for (int s = 0; s < nst; ++s) {
    const int cur = s & 1;
    if (s + 1 < nst) {
      issue_stage(s + 1, cur ^ 1);
      // 8 just-issued ops may remain in flight; async loads retire in order,
      // so <=8 outstanding guarantees stage s's 8 ops have landed in LDS.
      asm volatile("s_wait_asynccnt 0x8" ::: "memory");
    } else {
      asm volatile("s_wait_asynccnt 0x0" ::: "memory");
    }
    __syncthreads();  // publish buffer `cur` to all waves

    #pragma unroll
    for (int kk = 0; kk < KT; kk += 32) {
      v16bf a[4], b[2];
      #pragma unroll
      for (int i = 0; i < 4; ++i)
        a[i] = *reinterpret_cast<const v16bf*>(
            &As[cur][(wm * 64 + i * 16 + lrow) * KT + kk + lko]);
      #pragma unroll
      for (int j = 0; j < 2; ++j)
        b[j] = *reinterpret_cast<const v16bf*>(
            &Bs[cur][(wn * 32 + j * 16 + lrow) * KT + kk + lko]);
      #pragma unroll
      for (int i = 0; i < 4; ++i)
        #pragma unroll
        for (int j = 0; j < 2; ++j)
          acc[i][j] = __builtin_amdgcn_wmma_f32_16x16x32_bf16(
              false, a[i], false, b[j], (short)0, acc[i][j], false, false);
    }
    __syncthreads();  // all waves done reading `cur` before it is refilled
  }

  // Epilogue: D layout = VGPR r: row = r + (lane/16)*8, col = lane%16.
  const int mbase = m0 + wm * 64 + (lane >> 4) * 8;
  const int nbase = n0 + wn * 32 + (lane & 15);
  #pragma unroll
  for (int j = 0; j < 2; ++j) {
    const int col = nbase + j * 16;
    if (col >= nvalid) continue;
    const float bsum = (bias1 ? bias1[col] : 0.f) + (bias2 ? bias2[col] : 0.f);
    #pragma unroll
    for (int i = 0; i < 4; ++i) {
      #pragma unroll
      for (int r = 0; r < 8; ++r) {
        const int row = mbase + i * 16 + r;
        C[(long)row * ldc + col] = acc[i][j][r] + bsum;
      }
    }
  }
}

// ---------------------------------------------------------------------------
// LSTM pointwise: gates [B, 4H] (i,f,g,o) -> cn, hn (f32) + hn (bf16 for GEMM2)
// ---------------------------------------------------------------------------
__global__ void lstm_pointwise(const float* __restrict__ gates,
                               const float* __restrict__ c0,
                               float* __restrict__ hn_out, float* __restrict__ cn_out,
                               __bf16* __restrict__ hn_bf16) {
  const int idx = blockIdx.x * blockDim.x + threadIdx.x;
  if (idx >= B_DIM * H_DIM) return;
  const int b = idx / H_DIM;
  const int j = idx % H_DIM;
  const float* g = gates + (long)b * G_DIM;
  float ig = g[j];
  float fg = g[H_DIM + j];
  float gg = g[2 * H_DIM + j];
  float og = g[3 * H_DIM + j];
  ig = 1.f / (1.f + __expf(-ig));
  fg = 1.f / (1.f + __expf(-fg));
  gg = tanhf(gg);
  og = 1.f / (1.f + __expf(-og));
  const float c = fg * c0[idx] + ig * gg;
  const float h = og * tanhf(c);
  cn_out[idx]  = c;
  hn_out[idx]  = h;
  hn_bf16[idx] = (__bf16)h;
}

// ---------------------------------------------------------------------------
// Row softmax in place: one block per row, 256 threads (8 waves, wave32).
// ---------------------------------------------------------------------------
__global__ __launch_bounds__(256, 1)
void softmax_rows(float* __restrict__ logits, int ncols) {
  __shared__ float redm[8];
  __shared__ float reds[8];
  float* p = logits + (long)blockIdx.x * ncols;
  const int tid = threadIdx.x, lane = tid & 31, wave = tid >> 5;

  float m = -3.402823466e38f;
  for (int c = tid; c < ncols; c += 256) m = fmaxf(m, p[c]);
  #pragma unroll
  for (int o = 16; o > 0; o >>= 1) m = fmaxf(m, __shfl_xor(m, o, 32));
  if (lane == 0) redm[wave] = m;
  __syncthreads();
  m = redm[lane & 7];
  #pragma unroll
  for (int o = 4; o > 0; o >>= 1) m = fmaxf(m, __shfl_xor(m, o, 32));

  float s = 0.f;
  for (int c = tid; c < ncols; c += 256) s += __expf(p[c] - m);
  #pragma unroll
  for (int o = 16; o > 0; o >>= 1) s += __shfl_xor(s, o, 32);
  if (lane == 0) reds[wave] = s;
  __syncthreads();
  s = reds[lane & 7];
  #pragma unroll
  for (int o = 4; o > 0; o >>= 1) s += __shfl_xor(s, o, 32);

  const float inv = 1.f / s;
  for (int c = tid; c < ncols; c += 256) p[c] = __expf(p[c] - m) * inv;
}

// ---------------------------------------------------------------------------
extern "C" void kernel_launch(void* const* d_in, const int* in_sizes, int n_in,
                              void* d_out, int out_size, void* d_ws, size_t ws_size,
                              hipStream_t stream) {
  const float* input  = (const float*)d_in[0]; // [B, I]
  const float* hidden = (const float*)d_in[1]; // [1, B, H]
  const float* c_in   = (const float*)d_in[2]; // [1, B, H]
  const float* w_ih   = (const float*)d_in[3]; // [4H, I]
  const float* w_hh   = (const float*)d_in[4]; // [4H, H]
  const float* b_ih   = (const float*)d_in[5]; // [4H]
  const float* b_hh   = (const float*)d_in[6]; // [4H]
  const float* w_lin  = (const float*)d_in[7]; // [O, H]
  const float* b_lin  = (const float*)d_in[8]; // [O]

  float* out   = (float*)d_out;
  float* probs = out;                                   // [1, B, O]
  float* hn    = probs + (long)B_DIM * O_DIM;           // [1, B, H]
  float* cn    = hn + (long)B_DIM * H_DIM;              // [1, B, H]

  // Workspace layout (256B-aligned slabs).
  char* ws = (char*)d_ws;
  size_t off = 0;
  auto take = [&](size_t bytes) -> char* {
    char* p = ws + off;
    off += (bytes + 255) & ~(size_t)255;
    return p;
  };
  __bf16* w_lin_b = (__bf16*)take((size_t)O_PAD * H_DIM * 2);  // ~206 MB
  __bf16* w_ih_b  = (__bf16*)take((size_t)G_DIM * I_DIM * 2);  //  16 MB
  __bf16* w_hh_b  = (__bf16*)take((size_t)G_DIM * H_DIM * 2);  //  32 MB
  __bf16* in_b    = (__bf16*)take((size_t)B_DIM * I_DIM * 2);  //   2 MB
  __bf16* h0_b    = (__bf16*)take((size_t)B_DIM * H_DIM * 2);  //   4 MB
  __bf16* hn_b    = (__bf16*)take((size_t)B_DIM * H_DIM * 2);  //   4 MB
  float*  gates   = (float*) take((size_t)B_DIM * G_DIM * 4);  //  32 MB

  auto cvt = [&](const float* src, __bf16* dst, long n_src, long n_dst) {
    const long nthreads = n_dst / 4;
    const int blk = 256;
    const long grid = (nthreads + blk - 1) / blk;
    cvt_f32_to_bf16<<<(unsigned)grid, blk, 0, stream>>>(src, dst, n_src, n_dst);
  };
  cvt(input,  in_b,    (long)B_DIM * I_DIM, (long)B_DIM * I_DIM);
  cvt(hidden, h0_b,    (long)B_DIM * H_DIM, (long)B_DIM * H_DIM);
  cvt(w_ih,   w_ih_b,  (long)G_DIM * I_DIM, (long)G_DIM * I_DIM);
  cvt(w_hh,   w_hh_b,  (long)G_DIM * H_DIM, (long)G_DIM * H_DIM);
  cvt(w_lin,  w_lin_b, (long)O_DIM * H_DIM, (long)O_PAD * H_DIM); // zero-padded rows

  // Gates: [B, 4H] = in*w_ih^T + h0*w_hh^T + b_ih + b_hh  (fused dual-K GEMM)
  {
    dim3 grid(B_DIM / MT, G_DIM / NT); // (8, 64)
    gemm_bf16_nt<<<grid, 256, 0, stream>>>(in_b, w_ih_b, I_DIM,
                                           h0_b, w_hh_b, H_DIM,
                                           b_ih, b_hh,
                                           gates, (long)G_DIM, G_DIM);
  }

  // LSTM nonlinearity -> hn, cn, hn_bf16
  {
    const int n = B_DIM * H_DIM;
    lstm_pointwise<<<(n + 255) / 256, 256, 0, stream>>>(gates, c_in, hn, cn, hn_b);
  }

  // Logits: [B, O] = hn*w_lin^T + b_lin  (M-tile fastest -> w_lin strips reused in L2)
  {
    dim3 grid(B_DIM / MT, O_PAD / NT); // (8, 393)
    gemm_bf16_nt<<<grid, 256, 0, stream>>>(hn_b, w_lin_b, H_DIM,
                                           nullptr, nullptr, 0,
                                           b_lin, nullptr,
                                           probs, (long)O_DIM, O_DIM);
  }

  // Softmax in place over each row of probs.
  softmax_rows<<<B_DIM, 256, 0, stream>>>(probs, O_DIM);
}
